// MultiHeadSelfAttention_17987323036125
// MI455X (gfx1250) — matrix operand order repaired and checked by reading, not compiled, not run on hardware
//
#include <hip/hip_runtime.h>
#include <hip/hip_bf16.h>
#include <stdint.h>

#define EMBED  1024
#define NHEADS 16
#define HDIM   64
#define BATCH  4
#define SEQ    2048
#define MROWS  (BATCH*SEQ)   // 8192

typedef __attribute__((ext_vector_type(16))) __bf16        v16bf;
typedef __attribute__((ext_vector_type(8)))  float         v8f;
typedef __attribute__((ext_vector_type(8)))  unsigned int  v8u;

// ---------- helpers ----------

static __device__ __forceinline__ unsigned short f2bf(float f) {
    unsigned u = __builtin_bit_cast(unsigned, f);
    unsigned r = u + 0x7FFFu + ((u >> 16) & 1u);   // round-to-nearest-even
    return (unsigned short)(r >> 16);
}

static __device__ __forceinline__ v8f wmma_bf16(v16bf a, v16bf b, v8f c) {
    // (neg_a, A, neg_b, B, c_mod, C, reuse_a, reuse_b)
    return __builtin_amdgcn_wmma_f32_16x16x32_bf16(false, a, false, b, (short)0, c, false, false);
}

// A fragment: 16x32 bf16, row-major source, row stride ld (elements).
// Lane L: row = L&15, half = L>>4; VGPR v holds K = 16h+2v, pairs.
static __device__ __forceinline__ v16bf load_afrag(const unsigned short* base, int ld) {
    const int lane = threadIdx.x & 31;
    const int row  = lane & 15;
    const int half = lane >> 4;
    v8u u;
#pragma unroll
    for (int v = 0; v < 8; ++v) {
        const int kb = 16 * half + 2 * v;
        u[v] = *(const unsigned int*)(base + (size_t)row * ld + kb);
    }
    return __builtin_bit_cast(v16bf, u);
}

// B fragment: 32x16 bf16 (KxN). Source is "column-major by row" i.e. column n
// of B is a contiguous row of length K at base + n*ld (our transposed layouts).
// Lane L: col n = L&15, half = L>>4; VGPR v holds K = 16h+2v, pairs.
static __device__ __forceinline__ v16bf load_bfrag(const unsigned short* base, int ld) {
    const int lane = threadIdx.x & 31;
    const int col  = lane & 15;
    const int half = lane >> 4;
    v8u u;
#pragma unroll
    for (int v = 0; v < 8; ++v) {
        u[v] = *(const unsigned int*)(base + (size_t)col * ld + 16 * half + 2 * v);
    }
    return __builtin_bit_cast(v16bf, u);
}

// ---------- kernel 1: X f32 -> bf16 ----------

__global__ __launch_bounds__(256) void convert_x(const float* __restrict__ X,
                                                 unsigned short* __restrict__ Xbf,
                                                 int n4) {
    int i = blockIdx.x * 256 + threadIdx.x;
    if (i < n4) {
        float4 v = ((const float4*)X)[i];
        unsigned int lo = (unsigned)f2bf(v.x) | ((unsigned)f2bf(v.y) << 16);
        unsigned int hi = (unsigned)f2bf(v.z) | ((unsigned)f2bf(v.w) << 16);
        ((uint2*)Xbf)[i] = make_uint2(lo, hi);
    }
}

// ---------- kernel 2: transpose + convert weights (Wt[n][k] = W[k][n]) ----------

__global__ __launch_bounds__(256) void transpose_w(const float* __restrict__ wQ,
                                                   const float* __restrict__ wK,
                                                   const float* __restrict__ wV,
                                                   const float* __restrict__ wO,
                                                   unsigned short* __restrict__ wQt,
                                                   unsigned short* __restrict__ wKt,
                                                   unsigned short* __restrict__ wVt,
                                                   unsigned short* __restrict__ wOt) {
    __shared__ float tile[32][33];
    const int z = blockIdx.z;
    const float* src = (z == 0) ? wQ : (z == 1) ? wK : (z == 2) ? wV : wO;
    unsigned short* dst = (z == 0) ? wQt : (z == 1) ? wKt : (z == 2) ? wVt : wOt;

    const int x  = blockIdx.x * 32 + threadIdx.x;  // col n (coalesced read)
    const int y0 = blockIdx.y * 32;                // row k base
#pragma unroll
    for (int j = threadIdx.y; j < 32; j += 8)
        tile[j][threadIdx.x] = src[(size_t)(y0 + j) * EMBED + x];
    __syncthreads();
    const int n0 = blockIdx.x * 32;
#pragma unroll
    for (int j = threadIdx.y; j < 32; j += 8)
        dst[(size_t)(n0 + j) * EMBED + y0 + threadIdx.x] = f2bf(tile[threadIdx.x][j]);
}

// ---------- kernel 3: fused Q/K/V projection GEMM (bf16 WMMA, f32 acc) ----------
// One wave computes a 32x32 output tile. blockIdx.z selects Q/K/V.
// Q stored [B,H,S,64] pre-scaled by 1/8; K stored [B,H,S,64]; V stored [B,H,64,S].

__global__ __launch_bounds__(256) void gemm_qkv(const unsigned short* __restrict__ Xbf,
                                                const unsigned short* __restrict__ wQt,
                                                const unsigned short* __restrict__ wKt,
                                                const unsigned short* __restrict__ wVt,
                                                const float* __restrict__ bQ,
                                                const float* __restrict__ bK,
                                                const float* __restrict__ bV,
                                                unsigned short* __restrict__ Qbf,
                                                unsigned short* __restrict__ Kbf,
                                                unsigned short* __restrict__ Vt) {
    const int z = blockIdx.z;
    const unsigned short* Wt = (z == 0) ? wQt : (z == 1) ? wKt : wVt;
    const float* bias = (z == 0) ? bQ : (z == 1) ? bK : bV;

    const int wid  = threadIdx.x >> 5;
    const int lane = threadIdx.x & 31;
    const int colLane = lane & 15, half = lane >> 4;
    const int wg = blockIdx.x * 8 + wid;           // 8192 wave-jobs
    const int nt = wg & 31, mt = wg >> 5;          // 32 n-tiles, 256 m-tiles
    const int m0 = mt * 32, n0 = nt * 32;

    v8f acc[2][2] = {};
#pragma unroll 4
    for (int k0 = 0; k0 < EMBED; k0 += 32) {
        const unsigned short* abase = Xbf + (size_t)m0 * EMBED + k0;
        const unsigned short* bbase = Wt  + (size_t)n0 * EMBED + k0;
        __builtin_prefetch(abase + 32, 0, 1);      // global_prefetch next K chunk
        v16bf a0 = load_afrag(abase, EMBED);
        v16bf a1 = load_afrag(abase + (size_t)16 * EMBED, EMBED);
        v16bf b0 = load_bfrag(bbase, EMBED);
        v16bf b1 = load_bfrag(bbase + (size_t)16 * EMBED, EMBED);
        acc[0][0] = wmma_bf16(a0, b0, acc[0][0]);
        acc[0][1] = wmma_bf16(a0, b1, acc[0][1]);
        acc[1][0] = wmma_bf16(a1, b0, acc[1][0]);
        acc[1][1] = wmma_bf16(a1, b1, acc[1][1]);
    }

#pragma unroll
    for (int i = 0; i < 2; ++i) {
#pragma unroll
        for (int j = 0; j < 2; ++j) {
            const int nn = n0 + j * 16 + colLane;
            const float bval = bias[nn];
            const int head = nn >> 6, dd = nn & 63;
#pragma unroll
            for (int r = 0; r < 8; ++r) {
                const int m = m0 + i * 16 + r + 8 * half;
                const int bb = m >> 11, s = m & (SEQ - 1);
                const float val = acc[i][j][r] + bval;
                if (z == 0)
                    Qbf[((size_t)(bb * NHEADS + head) * SEQ + s) * HDIM + dd] = f2bf(val * 0.125f);
                else if (z == 1)
                    Kbf[((size_t)(bb * NHEADS + head) * SEQ + s) * HDIM + dd] = f2bf(val);
                else
                    Vt[((size_t)(bb * NHEADS + head) * HDIM + dd) * SEQ + s] = f2bf(val);
            }
        }
    }
}

// ---------- kernel 4: flash-style attention, one wave per 16-row q-tile ----------

__global__ __launch_bounds__(128) void attn_kernel(const unsigned short* __restrict__ Qbf,
                                                   const unsigned short* __restrict__ Kbf,
                                                   const unsigned short* __restrict__ Vt,
                                                   const int* __restrict__ mask,
                                                   unsigned short* __restrict__ attnOut) {
    __shared__ unsigned short Pbuf[4][16 * 32];    // 1 KB per wave: P tile re-layout
    const int wid  = threadIdx.x >> 5;
    const int lane = threadIdx.x & 31;
    const int colLane = lane & 15, half = lane >> 4;

    const int wg = blockIdx.x * 4 + wid;           // 8192 waves total
    const int qt = wg & 127;                       // 128 q-tiles per (b,h)
    const int bh = wg >> 7;                        // 0..63
    const int b  = bh >> 4;
    const int head = bh & 15;
    const int q0 = qt * 16;

    const unsigned short* qbase = Qbf + (size_t)(bh * SEQ + q0) * HDIM;
    const v16bf aQ0 = load_afrag(qbase, HDIM);        // qk dims 0..31 (pre-scaled by 1/8)
    const v16bf aQ1 = load_afrag(qbase + 32, HDIM);   // qk dims 32..63

    v8f o0 = {}, o1 = {}, o2 = {}, o3 = {};
    float m_run[8], l_run[8];
#pragma unroll
    for (int r = 0; r < 8; ++r) { m_run[r] = -3.0e38f; l_run[r] = 0.0f; }

    const int* mrow = mask + (size_t)b * SEQ * SEQ;
    unsigned short* Pw = &Pbuf[wid][0];

    for (int j0 = 0; j0 < SEQ; j0 += 32) {
        // ---- scores: Q(16x64) x K^T(64x32) = 4 WMMAs ----
        const unsigned short* kbase = Kbf + (size_t)(bh * SEQ + j0) * HDIM;
        v8f s0 = {}, s1 = {};
        s0 = wmma_bf16(aQ0, load_bfrag(kbase, HDIM), s0);
        s0 = wmma_bf16(aQ1, load_bfrag(kbase + 32, HDIM), s0);
        s1 = wmma_bf16(aQ0, load_bfrag(kbase + (size_t)16 * HDIM, HDIM), s1);
        s1 = wmma_bf16(aQ1, load_bfrag(kbase + (size_t)16 * HDIM + 32, HDIM), s1);

        // ---- mask (selects, no EXEC divergence) + online softmax ----
#pragma unroll
        for (int r = 0; r < 8; ++r) {
            const int q = q0 + r + 8 * half;
            const size_t mi = (size_t)q * SEQ + j0 + colLane;
            s0[r] = (mrow[mi]      == 0) ? -1e9f : s0[r];
            s1[r] = (mrow[mi + 16] == 0) ? -1e9f : s1[r];

            // row r+8h lives across the 16 lanes of this half-wave
            float t = fmaxf(s0[r], s1[r]);
            t = fmaxf(t, __shfl_xor(t, 1));
            t = fmaxf(t, __shfl_xor(t, 2));
            t = fmaxf(t, __shfl_xor(t, 4));
            t = fmaxf(t, __shfl_xor(t, 8));
            const float mnew  = fmaxf(m_run[r], t);
            const float alpha = __expf(m_run[r] - mnew);
            const float p0 = __expf(s0[r] - mnew);
            const float p1 = __expf(s1[r] - mnew);
            float rs = p0 + p1;
            rs += __shfl_xor(rs, 1);
            rs += __shfl_xor(rs, 2);
            rs += __shfl_xor(rs, 4);
            rs += __shfl_xor(rs, 8);
            l_run[r] = l_run[r] * alpha + rs;
            m_run[r] = mnew;
            o0[r] *= alpha; o1[r] *= alpha; o2[r] *= alpha; o3[r] *= alpha;

            // stage P in LDS (C-frag layout in, A-frag layout out)
            Pw[(r + 8 * half) * 32 + colLane]      = f2bf(p0);
            Pw[(r + 8 * half) * 32 + 16 + colLane] = f2bf(p1);
        }

        const v16bf aP = load_afrag(Pw, 32);

        // ---- PV: P(16x32) x V(32x64) = 4 WMMAs, V transposed so cols contiguous ----
        const unsigned short* vbase = Vt + (size_t)bh * HDIM * SEQ + j0;
        o0 = wmma_bf16(aP, load_bfrag(vbase, SEQ), o0);
        o1 = wmma_bf16(aP, load_bfrag(vbase + (size_t)16 * SEQ, SEQ), o1);
        o2 = wmma_bf16(aP, load_bfrag(vbase + (size_t)32 * SEQ, SEQ), o2);
        o3 = wmma_bf16(aP, load_bfrag(vbase + (size_t)48 * SEQ, SEQ), o3);
    }

    // ---- normalize + store bf16 into [B,S,H*64] for the output projection ----
#pragma unroll
    for (int r = 0; r < 8; ++r) {
        const float inv = 1.0f / l_run[r];
        const int q = q0 + r + 8 * half;
        unsigned short* dst = attnOut + (size_t)(b * SEQ + q) * EMBED + head * HDIM + colLane;
        dst[0]  = f2bf(o0[r] * inv);
        dst[16] = f2bf(o1[r] * inv);
        dst[32] = f2bf(o2[r] * inv);
        dst[48] = f2bf(o3[r] * inv);
    }
}

// ---------- kernel 5: output projection GEMM, f32 result + bias ----------

__global__ __launch_bounds__(256) void gemm_out(const unsigned short* __restrict__ Abf,
                                                const unsigned short* __restrict__ wOt,
                                                const float* __restrict__ bO,
                                                float* __restrict__ out) {
    const int wid  = threadIdx.x >> 5;
    const int lane = threadIdx.x & 31;
    const int colLane = lane & 15, half = lane >> 4;
    const int wg = blockIdx.x * 8 + wid;
    const int nt = wg & 31, mt = wg >> 5;
    const int m0 = mt * 32, n0 = nt * 32;

    v8f acc[2][2] = {};
#pragma unroll 4
    for (int k0 = 0; k0 < EMBED; k0 += 32) {
        const unsigned short* abase = Abf + (size_t)m0 * EMBED + k0;
        const unsigned short* bbase = wOt + (size_t)n0 * EMBED + k0;
        __builtin_prefetch(abase + 32, 0, 1);
        v16bf a0 = load_afrag(abase, EMBED);
        v16bf a1 = load_afrag(abase + (size_t)16 * EMBED, EMBED);
        v16bf b0 = load_bfrag(bbase, EMBED);
        v16bf b1 = load_bfrag(bbase + (size_t)16 * EMBED, EMBED);
        acc[0][0] = wmma_bf16(a0, b0, acc[0][0]);
        acc[0][1] = wmma_bf16(a0, b1, acc[0][1]);
        acc[1][0] = wmma_bf16(a1, b0, acc[1][0]);
        acc[1][1] = wmma_bf16(a1, b1, acc[1][1]);
    }
#pragma unroll
    for (int i = 0; i < 2; ++i) {
#pragma unroll
        for (int j = 0; j < 2; ++j) {
            const int nn = n0 + j * 16 + colLane;
            const float bval = bO[nn];
#pragma unroll
            for (int r = 0; r < 8; ++r) {
                const int m = m0 + i * 16 + r + 8 * half;
                out[(size_t)m * EMBED + nn] = acc[i][j][r] + bval;
            }
        }
    }
}

// ---------- host launcher ----------

extern "C" void kernel_launch(void* const* d_in, const int* in_sizes, int n_in,
                              void* d_out, int out_size, void* d_ws, size_t ws_size,
                              hipStream_t stream) {
    (void)in_sizes; (void)n_in; (void)out_size; (void)ws_size;
    const float* X    = (const float*)d_in[0];
    const int*   mask = (const int*)d_in[1];
    const float* wQ = (const float*)d_in[2];
    const float* bQ = (const float*)d_in[3];
    const float* wK = (const float*)d_in[4];
    const float* bK = (const float*)d_in[5];
    const float* wV = (const float*)d_in[6];
    const float* bV = (const float*)d_in[7];
    const float* wO = (const float*)d_in[8];
    const float* bO = (const float*)d_in[9];
    float* out = (float*)d_out;

    char* ws = (char*)d_ws;
    size_t off = 0;
    auto alloc = [&](size_t bytes) {
        char* p = ws + off;
        off += (bytes + 255) & ~(size_t)255;
        return p;
    };
    unsigned short* Xbf = (unsigned short*)alloc((size_t)MROWS * EMBED * 2);   // 16 MB
    unsigned short* wQt = (unsigned short*)alloc((size_t)EMBED * EMBED * 2);   //  2 MB
    unsigned short* wKt = (unsigned short*)alloc((size_t)EMBED * EMBED * 2);
    unsigned short* wVt = (unsigned short*)alloc((size_t)EMBED * EMBED * 2);
    unsigned short* wOt = (unsigned short*)alloc((size_t)EMBED * EMBED * 2);
    unsigned short* Qbf = (unsigned short*)alloc((size_t)MROWS * EMBED * 2);   // [B,H,S,64]
    unsigned short* Kbf = (unsigned short*)alloc((size_t)MROWS * EMBED * 2);   // [B,H,S,64]
    unsigned short* Vt  = (unsigned short*)alloc((size_t)MROWS * EMBED * 2);   // [B,H,64,S]
    unsigned short* attnOut = Xbf;  // reuse: Xbf dead after gemm_qkv (stream-ordered)

    convert_x<<<(MROWS * EMBED / 4 + 255) / 256, 256, 0, stream>>>(X, Xbf, MROWS * EMBED / 4);
    transpose_w<<<dim3(32, 32, 4), dim3(32, 8), 0, stream>>>(wQ, wK, wV, wO, wQt, wKt, wVt, wOt);
    gemm_qkv<<<dim3(1024, 1, 3), 256, 0, stream>>>(Xbf, wQt, wKt, wVt, bQ, bK, bV, Qbf, Kbf, Vt);
    attn_kernel<<<2048, 128, 0, stream>>>(Qbf, Kbf, Vt, mask, attnOut);
    gemm_out<<<1024, 256, 0, stream>>>(attnOut, wOt, bO, out);
}
